// SpectralMultiheadAttention_25391846654343
// MI455X (gfx1250) — matrix operand order repaired and checked by reading, not compiled, not run on hardware
//
#include <hip/hip_runtime.h>

typedef __attribute__((ext_vector_type(16))) _Float16 v16h;
typedef __attribute__((ext_vector_type(8)))  float    v8f;

#define WMMA_F16(a, b, c) \
  __builtin_amdgcn_wmma_f32_16x16x32_f16(false, (a), false, (b), (short)0, (c), false, false)

union Frag16 { v16h v; _Float16 h[16]; uint4 q[2]; };
union Half8  { uint4 q; _Float16 h[8]; };
union Half4  { uint2 u; _Float16 h[4]; };

constexpr int Tt  = 64;    // BLOCK (sequence length)
constexpr int Cc  = 128;   // N_EMBED
constexpr int Hh  = 8;     // NUM_HEADS
constexpr int Dd  = 16;    // HEAD_SIZE
constexpr int XLD = 136;   // LDS stride (halves) for x / concat tile (272B rows, 16B aligned)
constexpr int QLD = 16;    // LDS stride (halves) for Q/K (32B rows, 16B aligned)
constexpr int VLD = 72;    // LDS stride (halves) for V^T (144B rows, 16B aligned)
constexpr int PLD = 72;    // LDS stride (halves) for P (144B rows, 16B aligned)

// ---------------- WMMA fragment helpers (layouts per CDNA5 ISA 7.12.2) ----------------

// A-matrix 16x32 f16 from row-major [16][ld] halves; 2x ds_load_b128 per lane.
__device__ inline v16h load_a_lds(const _Float16* base, int ld, int lane) {
  const int row  = lane & 15;
  const int koff = (lane < 16) ? 0 : 16;
  const _Float16* p = base + row * ld + koff;
  Frag16 f;
  f.q[0] = *(const uint4*)(p);        // halves 0..7   (K = koff+0..7)
  f.q[1] = *(const uint4*)(p + 8);    // halves 8..15  (K = koff+8..15)
  return f.v;
}

// A-matrix with real K=16 zero-padded to 32; 1x ds_load_b128 per lane.
__device__ inline v16h load_a_pad16(const _Float16* base, int ld, int lane) {
  const int row  = lane & 15;
  const int koff = (lane < 16) ? 0 : 8;
  Frag16 f;
  f.q[0] = *(const uint4*)(base + row * ld + koff);
  f.q[1] = uint4{0u, 0u, 0u, 0u};
  return f.v;
}

// B = M^T (16 valid K rows padded to 32) from row-major M tile [16][ld]; lane half h reads K = 8h..8h+7.
__device__ inline v16h load_bT_pad16(const _Float16* m, int ld, int lane) {
  Frag16 f;
  {
    const _Float16* p = m + (lane & 15) * ld + ((lane < 16) ? 0 : 8);
    f.q[0] = *(const uint4*)(p);
    f.q[1] = uint4{0u, 0u, 0u, 0u};
  }
  return f.v;
}

// Generic B-fragment from an [n][ld] row-major array (k contiguous within a row).
__device__ inline v16h load_b_rows(const _Float16* base, int ld, int koffset, int lane) {
  const int col = lane & 15;
  const int kb  = (lane < 16) ? 0 : 8;
  const _Float16* p = base + col * ld + koffset + kb;
  Frag16 f;
  f.q[0] = *(const uint4*)(p);
  f.q[1] = *(const uint4*)(p + 16);
  return f.v;
}

// C/D f32 tile -> f16 row-major [16][ld] (scalar b16 stores).
__device__ inline void store_c_f16(const v8f& c, _Float16* base, int ld, int lane) {
  const int n  = lane & 15;
  const int mb = (lane < 16) ? 0 : 8;
#pragma unroll
  for (int i = 0; i < 8; ++i) base[(mb + i) * ld + n] = (_Float16)c[i];
}

// C/D f32 tile -> f16 TRANSPOSED [n][ld]: one packed b128 store per lane.
__device__ inline void store_c_f16_T(const v8f& c, _Float16* base, int ld, int lane) {
  const int n  = lane & 15;
  const int mb = (lane < 16) ? 0 : 8;
  Half8 t;
#pragma unroll
  for (int i = 0; i < 8; ++i) t.h[i] = (_Float16)c[i];
  *(uint4*)(base + n * ld + mb) = t.q;
}

// ---------------- weight prep ----------------
// wq/wk/wv: B-fragment layout frag[(h*4+kt)*32+lane][16]; element j = W[kt*32+kb+j][col].
// wq additionally absorbs the 128^-0.5 attention scale.
// pwp: proj_w^T packed as A-fragments frag[(g*4+kt)*32+lane][16] for out^T = pw^T @ concat.

__global__ __launch_bounds__(256) void prep_weights(
    const float* __restrict__ qc, const float* __restrict__ kc, const float* __restrict__ vc,
    const float* __restrict__ qu, const float* __restrict__ ku, const float* __restrict__ vu,
    const float* __restrict__ pwf,
    _Float16* __restrict__ wqp, _Float16* __restrict__ wkp, _Float16* __restrict__ wvp,
    _Float16* __restrict__ pwp) {
  const int i    = blockIdx.x * 256 + threadIdx.x;  // 0 .. 16383 (packed linear index)
  const int j    = i & 15;
  const int lane = (i >> 4) & 31;
  const int kt   = (i >> 9) & 3;
  const int g    = (i >> 11) & 7;

  // ---- B-fragment decode (Q/K/V weights) ----
  const int colB = lane & 15;
  const int kbB  = (lane < 16) ? 0 : 16;
  const int rowB = kt * 32 + kbB + j;               // 0..127 (embedding dim)
  const int wsrc = rowB * Dd + colB;
  wqp[i] = (_Float16)((qc[wsrc] + qu[g * (Cc * Dd) + wsrc]) * 0.088388347648318447f);
  wkp[i] = (_Float16)(kc[wsrc] + ku[g * (Cc * Dd) + wsrc]);
  wvp[i] = (_Float16)(vc[wsrc] + vu[g * (Cc * Dd) + wsrc]);

  // ---- A-fragment decode (proj_w^T): row m = c_out local, K per ISA A layout ----
  const int mA    = lane & 15;                      // c_out within tile
  const int koffA = (lane < 16) ? 0 : 8;
  const int jj    = j >> 1, pr = j & 1;
  const int kmap  = (jj < 4) ? (2 * jj) : (8 + 2 * jj);
  const int kA    = kt * 32 + koffA + kmap + pr;    // c_in 0..127
  pwp[i] = (_Float16)pwf[kA * Cc + g * 16 + mA];
}

// ---------------- fused attention: one block per batch element, one wave per head ----------------

__global__ __launch_bounds__(256) void attn_fused(
    const float* __restrict__ x,
    const _Float16* __restrict__ wqp, const _Float16* __restrict__ wkp,
    const _Float16* __restrict__ wvp, const _Float16* __restrict__ pwp,
    const float* __restrict__ projb,
    float* __restrict__ out) {
  __shared__ __align__(16) _Float16 xh[Tt * XLD];            // x tile f16; reused as concat buffer
  __shared__ __align__(16) _Float16 qk[Hh][2][Tt * QLD];     // Q,K row-major
  __shared__ __align__(16) _Float16 vt[Hh][Dd * VLD];        // V^T
  __shared__ __align__(16) _Float16 pbuf[Hh][Tt * PLD];      // P row-major [t][s]

  const int b    = blockIdx.x;
  const int tid  = threadIdx.x;
  const int wave = tid >> 5;                                 // head id
  const int lane = tid & 31;
  const int n    = lane & 15;
  const int mb   = (lane < 16) ? 0 : 8;

  // ---- stage x[b] -> LDS f16 (float4 loads, packed 8B LDS stores) ----
  {
    const float4* xb4 = (const float4*)(x + (size_t)b * (Tt * Cc));
#pragma unroll
    for (int it = 0; it < (Tt * Cc / 4) / 256; ++it) {
      const int i4 = tid + it * 256;        // 0..2047
      const float4 vx = xb4[i4];
      const int r = i4 >> 5;
      const int c = (i4 & 31) * 4;
      Half4 t;
      t.h[0] = (_Float16)vx.x; t.h[1] = (_Float16)vx.y;
      t.h[2] = (_Float16)vx.z; t.h[3] = (_Float16)vx.w;
      *(uint2*)(xh + r * XLD + c) = t.u;
    }
  }
  __syncthreads();

  // ---- Q,K,V = x @ W_eff[head]  (scale folded into Wq) ----
  {
    const int fo = wave * 2048 + lane * 16;
    v16h bq[4], bk[4], bv[4];
#pragma unroll
    for (int kt = 0; kt < 4; ++kt) {
      bq[kt] = *(const v16h*)(wqp + fo + kt * 512);
      bk[kt] = *(const v16h*)(wkp + fo + kt * 512);
      bv[kt] = *(const v16h*)(wvp + fo + kt * 512);
    }
#pragma unroll
    for (int mt = 0; mt < 4; ++mt) {
      v16h a[4];
#pragma unroll
      for (int kt = 0; kt < 4; ++kt)
        a[kt] = load_a_lds(xh + mt * 16 * XLD + kt * 32, XLD, lane);
      v8f cq = {}, ck = {}, cv = {};
#pragma unroll
      for (int kt = 0; kt < 4; ++kt) {
        cq = WMMA_F16(a[kt], bq[kt], cq);
        ck = WMMA_F16(a[kt], bk[kt], ck);
        cv = WMMA_F16(a[kt], bv[kt], cv);
      }
      store_c_f16(cq, qk[wave][0] + mt * 16 * QLD, QLD, lane);
      store_c_f16(ck, qk[wave][1] + mt * 16 * QLD, QLD, lane);
      store_c_f16_T(cv, vt[wave] + mt * 16, VLD, lane);      // V^T [d][t]
    }
  }
  __syncthreads();  // all waves done reading xh -> becomes concat buffer

  // ---- causal attention, computed TRANSPOSED: S^T = K @ Q^T ----
  // C layout of S^T tile (st,tt): lane -> t = tt*16+n (column), regs i -> s = st*16+mb+i.
  {
    const _Float16* Q = qk[wave][0];
    const _Float16* K = qk[wave][1];
    _Float16* P = pbuf[wave];

    v16h bqf[4], aKf[4];
#pragma unroll
    for (int tt = 0; tt < 4; ++tt) bqf[tt] = load_bT_pad16(Q + tt * 16 * QLD, QLD, lane);
#pragma unroll
    for (int st = 0; st < 4; ++st) aKf[st] = load_a_pad16(K + st * 16 * QLD, QLD, lane);

#pragma unroll
    for (int tt = 0; tt < 4; ++tt) {
      v8f sT[4] = {};
#pragma unroll
      for (int st = 0; st < 4; ++st) {
        if (st <= tt) {
          v8f z = {};
          sT[st] = WMMA_F16(aKf[st], bqf[tt], z);
        }
      }
      // softmax over s (within-lane regs + one cross-half shuffle); no max-sub (|logit| <~ 1)
      const int t = tt * 16 + n;
      float pv[4][8];
      float psum = 0.f;
#pragma unroll
      for (int st = 0; st < 4; ++st) {
        if (st <= tt) {
#pragma unroll
          for (int i = 0; i < 8; ++i) {
            const int s = st * 16 + mb + i;
            const float p = (s <= t) ? __expf(sT[st][i]) : 0.f;   // causal mask
            pv[st][i] = p;
            psum += p;
          }
        }
      }
      psum += __shfl_xor(psum, 16, 32);     // combine the two lane-halves of column t
      const float inv = 1.f / psum;
      // store P[t][s] (normalized), zero-filling s-tiles above the diagonal: b128 per tile
      _Float16* prow = P + t * PLD + mb;
#pragma unroll
      for (int st = 0; st < 4; ++st) {
        Half8 t8;
        if (st <= tt) {
#pragma unroll
          for (int i = 0; i < 8; ++i) t8.h[i] = (_Float16)(pv[st][i] * inv);
        } else {
          t8.q = uint4{0u, 0u, 0u, 0u};
        }
        *(uint4*)(prow + st * 16) = t8.q;
      }
    }

    // out_head = P @ V -> concat buffer (xh) columns [wave*16, wave*16+16)
    v16h bvf[2];
#pragma unroll
    for (int kt = 0; kt < 2; ++kt) {
      const _Float16* p = vt[wave] + n * VLD + kt * 32 + ((lane < 16) ? 0 : 16);
      Frag16 f;
      f.q[0] = *(const uint4*)(p);
      f.q[1] = *(const uint4*)(p + 8);
      bvf[kt] = f.v;
    }
#pragma unroll
    for (int mt = 0; mt < 4; ++mt) {
      v8f acc = {};
#pragma unroll
      for (int kt = 0; kt < 2; ++kt) {
        const v16h ap = load_a_lds(P + mt * 16 * PLD + kt * 32, PLD, lane);
        acc = WMMA_F16(ap, bvf[kt], acc);
      }
      store_c_f16(acc, xh + mt * 16 * XLD + wave * 16, XLD, lane);
    }
  }
  __syncthreads();

  // ---- out^T = proj_w^T @ concat (+bias); wave owns c_out tile `wave` ----
  // C layout: lane -> t = tt*16+n, regs i -> c_out = wave*16+mb+i (contiguous -> b128 stores)
  {
    const int cbase = wave * 16 + mb;
    float4 bia0 = *(const float4*)(projb + cbase);
    float4 bia1 = *(const float4*)(projb + cbase + 4);
    float* ob = out + (size_t)b * (Tt * Cc);
    const int fo = wave * 2048 + lane * 16;
    v16h apf[4];
#pragma unroll
    for (int kt = 0; kt < 4; ++kt) apf[kt] = *(const v16h*)(pwp + fo + kt * 512);
#pragma unroll
    for (int tt = 0; tt < 4; ++tt) {
      v8f acc = {};
#pragma unroll
      for (int kt = 0; kt < 4; ++kt) {
        const v16h bc = load_b_rows(xh + tt * 16 * XLD, XLD, kt * 32, lane);
        acc = WMMA_F16(apf[kt], bc, acc);
      }
      float* dst = ob + (tt * 16 + n) * Cc + cbase;
      float4 lo = make_float4(acc[0] + bia0.x, acc[1] + bia0.y, acc[2] + bia0.z, acc[3] + bia0.w);
      float4 hi = make_float4(acc[4] + bia1.x, acc[5] + bia1.y, acc[6] + bia1.z, acc[7] + bia1.w);
      *(float4*)(dst)     = lo;
      *(float4*)(dst + 4) = hi;
    }
  }
}

// ---------------- host launcher ----------------

extern "C" void kernel_launch(void* const* d_in, const int* in_sizes, int n_in,
                              void* d_out, int out_size, void* d_ws, size_t ws_size,
                              hipStream_t stream) {
  (void)n_in; (void)out_size; (void)ws_size;
  const float* x     = (const float*)d_in[0];
  const float* Wq_c  = (const float*)d_in[1];
  const float* Wk_c  = (const float*)d_in[2];
  const float* Wv_c  = (const float*)d_in[3];
  const float* Wq_u  = (const float*)d_in[4];
  const float* Wk_u  = (const float*)d_in[5];
  const float* Wv_u  = (const float*)d_in[6];
  const float* projw = (const float*)d_in[7];
  const float* projb = (const float*)d_in[8];
  float* out = (float*)d_out;

  // workspace: 4x 16384 f16 fragment-packed weight arrays = 128 KiB
  _Float16* wsh = (_Float16*)d_ws;
  _Float16* wqp = wsh;
  _Float16* wkp = wsh + Hh * Cc * Dd;
  _Float16* wvp = wsh + 2 * Hh * Cc * Dd;
  _Float16* pwp = wsh + 3 * Hh * Cc * Dd;

  prep_weights<<<(Hh * Cc * Dd) / 256, 256, 0, stream>>>(
      Wq_c, Wk_c, Wv_c, Wq_u, Wk_u, Wv_u, projw, wqp, wkp, wvp, pwp);

  const int B = in_sizes[0] / (Tt * Cc);  // 4096
  attn_fused<<<B, 256, 0, stream>>>(x, wqp, wkp, wvp, pwp, projb, out);
}
